// QKMClassModel_2568390443651
// MI455X (gfx1250) — matrix operand order repaired and checked by reading, not compiled, not run on hardware
//
#include <hip/hip_runtime.h>
#include <hip/hip_bf16.h>

// ---------------------------------------------------------------------------
// QKM classifier pipeline for MI455X (gfx1250, wave32, WMMA).
//   enc   = relu(x @ W + b)                      4096x1024 * 1024x512
//   G     = enc @ c_x^T                          4096x512  * 512x4096
//   d2    = |enc|^2 + |c|^2 - 2G ; kern = exp(-max(d2,0)/(2 sigma^2))
//   w     = kern^2 * c_w ; probs = (w @ y^2) / rowsum(w),  y = normalize(c_y)
// Both GEMMs run on v_wmma_f32_16x16x32_f16 (f16 in, f32 accum). GEMM tile
// staging uses CDNA5 async global->LDS copies (GLOBAL_LOAD_ASYNC_TO_LDS_B128,
// ASYNCcnt) instead of a VGPR round-trip. The BxM weight matrix is never
// written to HBM: the RBF weighting and the tiny w @ y^2 (DY=10) contraction
// are fused into GEMM2's epilogue via LDS.
// ---------------------------------------------------------------------------

typedef _Float16 half_t;
typedef __attribute__((ext_vector_type(16))) _Float16 v16h;
typedef __attribute__((ext_vector_type(8)))  float    v8f;

static constexpr int Bn    = 4096;
static constexpr int D_IN  = 1024;
static constexpr int D_ENC = 512;
static constexpr int Mn    = 4096;
static constexpr int DY    = 10;

static constexpr int LDSP  = 40;   // padded LDS stride in halves; 80B = 5*16B

// --------------------- CDNA5 async global->LDS helpers ---------------------
// the LDS destination address is passed in the vdst VGPR; low 32 bits of a __shared__ generic pointer
// are the allocation-relative LDS byte offset. Tracked with ASYNCcnt.
__device__ __forceinline__ unsigned lds_off(const void* p) {
  return (unsigned)(size_t)p;
}
__device__ __forceinline__ void async_copy_b128(unsigned lds, const void* g) {
  asm volatile("global_load_async_to_lds_b128 %0, %1, off"
               :: "v"(lds), "v"(g) : "memory");
}
__device__ __forceinline__ void wait_async0() {
  asm volatile("s_wait_asynccnt 0" ::: "memory");
}

// ---------------- fragment loaders (contiguous-K order, wave32) ------------
// A fragment 16x32 f16: lane 0-15 -> row M=lane, K in {0..15};
//                       lane 16-31 -> row M=lane-16, K in {16..31}.
__device__ inline v16h load_A_frag(const half_t* lds, int row0, int lane) {
  v16h a;
  const half_t* p = lds + (size_t)(row0 + (lane & 15)) * LDSP;
  const int kb = (lane < 16) ? 0 : 16;
#pragma unroll
  for (int v = 0; v < 8; ++v) {
    const int k = 2 * v + kb;   // pairs k,k+1
    a[2 * v]     = p[k];
    a[2 * v + 1] = p[k + 1];
  }
  return a;
}

// B fragment 32x16 f16 from an N-major (col-of-B contiguous-in-K) LDS tile:
// lane 0-15 hold K=0..15 (2 per VGPR), lane 16-31 hold K=16..31; N = lane&15.
__device__ inline v16h load_B_frag(const half_t* lds, int col0, int lane) {
  v16h b;
  const half_t* p = lds + (size_t)(col0 + (lane & 15)) * LDSP;
  const int kb = (lane < 16) ? 0 : 16;
#pragma unroll
  for (int v = 0; v < 8; ++v) {
    b[2 * v]     = p[kb + 2 * v];
    b[2 * v + 1] = p[kb + 2 * v + 1];
  }
  return b;
}

__device__ inline v8f wmma_f16(v16h a, v16h b, v8f c) {
  return __builtin_amdgcn_wmma_f32_16x16x32_f16(false, a, false, b,
                                                (short)0, c, false, false);
}

// ------------------------------ prep kernels -------------------------------
__global__ void cvt_f16_kernel(const float* __restrict__ s,
                               half_t* __restrict__ d, int n) {
  int i = blockIdx.x * 256 + threadIdx.x;
  if (i < n) d[i] = (half_t)s[i];
}

// W[K x N] (row-major) -> Wt[N x K] f16 (N-major) so GEMM1 B-staging is
// contiguous in K, same as c_x for GEMM2.
__global__ void cvt_f16_T_kernel(const float* __restrict__ s,
                                 half_t* __restrict__ d, int K, int N) {
  int i = blockIdx.x * 256 + threadIdx.x;
  if (i < K * N) {
    int n = i / K, k = i - n * K;
    d[i] = (half_t)s[(size_t)k * N + n];
  }
}

__global__ void row_norm2_f32_kernel(const float* __restrict__ src,
                                     float* __restrict__ out, int ncols) {
  const int row = blockIdx.x, lane = threadIdx.x;   // blockDim == 32 (1 wave)
  const float* p = src + (size_t)row * ncols;
  float s = 0.f;
  for (int c = lane; c < ncols; c += 32) { float v = p[c]; s += v * v; }
#pragma unroll
  for (int off = 16; off > 0; off >>= 1) s += __shfl_down(s, off, 32);
  if (lane == 0) out[row] = s;
}

__global__ void row_norm2_f16_kernel(const half_t* __restrict__ src,
                                     float* __restrict__ out, int ncols) {
  const int row = blockIdx.x, lane = threadIdx.x;
  const half_t* p = src + (size_t)row * ncols;
  float s = 0.f;
  for (int c = lane; c < ncols; c += 32) { float v = (float)p[c]; s += v * v; }
#pragma unroll
  for (int off = 16; off > 0; off >>= 1) s += __shfl_down(s, off, 32);
  if (lane == 0) out[row] = s;
}

// y = c_y / ||c_y|| ; y2 = y*y  (note (v/||.||)^2 = v^2 / sum(v^2), no sqrt)
__global__ void prep_y2_kernel(const float* __restrict__ cy,
                               float* __restrict__ y2, int m) {
  int i = blockIdx.x * 256 + threadIdx.x;
  if (i < m) {
    const float* p = cy + (size_t)i * DY;
    float s = 0.f;
#pragma unroll
    for (int d = 0; d < DY; ++d) { float v = p[d]; s += v * v; }
    const float inv = 1.0f / s;
#pragma unroll
    for (int d = 0; d < DY; ++d) { float v = p[d]; y2[(size_t)i * DY + d] = v * v * inv; }
  }
}

// ------------------------- GEMM1: enc = relu(xW+b) -------------------------
// 128x128 block tile, 256 threads = 8 waves (4x2), each wave 32x64 = 2x4 frags.
__global__ __launch_bounds__(256)
void gemm_enc_kernel(const half_t* __restrict__ X,    // Bn x D_IN, row-major
                     const half_t* __restrict__ Wt,   // D_ENC x D_IN, N-major
                     const float*  __restrict__ bias, // D_ENC
                     half_t* __restrict__ enc) {      // Bn x D_ENC
  __shared__ __align__(16) half_t As[128 * LDSP];
  __shared__ __align__(16) half_t Bs[128 * LDSP];

  const int tid  = threadIdx.x;
  const int lane = tid & 31;
  const int wave = tid >> 5;
  const int wr   = wave >> 1;          // 0..3  -> 32-row strips
  const int wc   = wave & 1;           // 0..1  -> 64-col strips
  const int row_blk = blockIdx.x * 128;
  const int col_blk = blockIdx.y * 128;

  v8f acc[2][4] = {};

  for (int k0 = 0; k0 < D_IN; k0 += 32) {
    {  // async stage: 128 rows x 64B each side; 2 threads/row, 2x16B/thread
      const int r = tid >> 1, seg = (tid & 1) * 16;
      const half_t* gx = X  + (size_t)(row_blk + r) * D_IN + k0 + seg;
      const half_t* gw = Wt + (size_t)(col_blk + r) * D_IN + k0 + seg;
      half_t* sa = As + (size_t)r * LDSP + seg;
      half_t* sb = Bs + (size_t)r * LDSP + seg;
      async_copy_b128(lds_off(sa),     gx);
      async_copy_b128(lds_off(sa + 8), gx + 8);
      async_copy_b128(lds_off(sb),     gw);
      async_copy_b128(lds_off(sb + 8), gw + 8);
    }
    wait_async0();
    __syncthreads();
    v16h a0 = load_A_frag(As, wr * 32 + 0,  lane);
    v16h a1 = load_A_frag(As, wr * 32 + 16, lane);
#pragma unroll
    for (int fc = 0; fc < 4; ++fc) {
      v16h b = load_B_frag(Bs, wc * 64 + fc * 16, lane);
      acc[0][fc] = wmma_f16(a0, b, acc[0][fc]);
      acc[1][fc] = wmma_f16(a1, b, acc[1][fc]);
    }
    __syncthreads();
  }

  // epilogue: C/D layout -> row = v (+8 for upper half-wave), col = lane&15
  const int col16  = lane & 15;
  const int rowadd = (lane < 16) ? 0 : 8;
#pragma unroll
  for (int fr = 0; fr < 2; ++fr)
#pragma unroll
    for (int fc = 0; fc < 4; ++fc) {
      const int col = col_blk + wc * 64 + fc * 16 + col16;
      const float bv = bias[col];
#pragma unroll
      for (int v = 0; v < 8; ++v) {
        const int row = row_blk + wr * 32 + fr * 16 + v + rowadd;
        float val = acc[fr][fc][v] + bv;
        val = fmaxf(val, 0.0f);
        enc[(size_t)row * D_ENC + col] = (half_t)val;
      }
    }
}

// ---------------- GEMM2 fused: RBF kernel + weighting + probs --------------
// 64-row block, 128 threads = 4 waves (2x2), each wave 32x64 = 2x4 frags.
// Loops over all M in 128-col tiles; w-tile lives only in LDS.
__global__ __launch_bounds__(128)
void gemm_qkm_kernel(const half_t* __restrict__ Enc,    // Bn x D_ENC
                     const half_t* __restrict__ Cx,     // Mn x D_ENC (N-major B)
                     const float*  __restrict__ enc_n2, // Bn
                     const float*  __restrict__ c_n2,   // Mn
                     const float*  __restrict__ c_w,    // Mn
                     const float*  __restrict__ y2,     // Mn x DY
                     const float*  __restrict__ sigma,  // 1
                     float* __restrict__ out) {         // Bn x DY
  __shared__ __align__(16) half_t As[64 * LDSP];   //  5.0 KB
  __shared__ __align__(16) half_t Bs[128 * LDSP];  // 10.0 KB
  __shared__ float  wt[64 * 128];                  // 32.0 KB
  __shared__ float  pacc[64 * DY];                 //  2.5 KB
  __shared__ float  rsum[64];                      //  0.25 KB (total < 64 KB)

  const int tid  = threadIdx.x;
  const int lane = tid & 31;
  const int wave = tid >> 5;
  const int wr   = wave >> 1;            // 0..1 -> 32-row strips
  const int wc   = wave & 1;             // 0..1 -> 64-col strips
  const int row_blk = blockIdx.x * 64;

  const float sg = sigma[0];
  const float s2inv = 1.0f / (2.0f * sg * sg);

  for (int i = tid; i < 64 * DY; i += 128) pacc[i] = 0.0f;
  if (tid < 64) rsum[tid] = 0.0f;
  __syncthreads();

  const int col16  = lane & 15;
  const int rowadd = (lane < 16) ? 0 : 8;

  for (int mt = 0; mt < Mn; mt += 128) {
    v8f acc[2][4] = {};

    for (int k0 = 0; k0 < D_ENC; k0 += 32) {
      {  // async stage A: 64 rows, 2 threads/row; B: 128 rows, 1 thread/row
        const int r = tid >> 1, seg = (tid & 1) * 16;
        const half_t* ge = Enc + (size_t)(row_blk + r) * D_ENC + k0 + seg;
        half_t* sa = As + (size_t)r * LDSP + seg;
        async_copy_b128(lds_off(sa),     ge);
        async_copy_b128(lds_off(sa + 8), ge + 8);

        const half_t* gc = Cx + (size_t)(mt + tid) * D_ENC + k0;
        half_t* sb = Bs + (size_t)tid * LDSP;
        async_copy_b128(lds_off(sb),      gc);
        async_copy_b128(lds_off(sb + 8),  gc + 8);
        async_copy_b128(lds_off(sb + 16), gc + 16);
        async_copy_b128(lds_off(sb + 24), gc + 24);
      }
      wait_async0();
      __syncthreads();
      v16h a0 = load_A_frag(As, wr * 32 + 0,  lane);
      v16h a1 = load_A_frag(As, wr * 32 + 16, lane);
#pragma unroll
      for (int fc = 0; fc < 4; ++fc) {
        v16h b = load_B_frag(Bs, wc * 64 + fc * 16, lane);
        acc[0][fc] = wmma_f16(a0, b, acc[0][fc]);
        acc[1][fc] = wmma_f16(a1, b, acc[1][fc]);
      }
      __syncthreads();
    }

    // fragment -> w tile in LDS: w = exp(-max(d2,0)/(2s^2))^2 * c_w
#pragma unroll
    for (int fr = 0; fr < 2; ++fr)
#pragma unroll
      for (int fc = 0; fc < 4; ++fc) {
        const int col_l = wc * 64 + fc * 16 + col16;
        const float cn  = c_n2[mt + col_l];
        const float cwv = c_w[mt + col_l];
#pragma unroll
        for (int v = 0; v < 8; ++v) {
          const int row_l = wr * 32 + fr * 16 + v + rowadd;
          const float g  = acc[fr][fc][v];
          float d2 = enc_n2[row_blk + row_l] + cn - 2.0f * g;
          d2 = fmaxf(d2, 0.0f);
          const float kern = __expf(-d2 * s2inv);
          wt[row_l * 128 + col_l] = kern * kern * cwv;
        }
      }
    __syncthreads();

    // pacc[row][dy] += sum_c wt[row][c] * y2[mt+c][dy];  rsum[row] += sum_c wt
    for (int idx = tid; idx < 64 * DY; idx += 128) {
      const int row = idx / DY, dy = idx - row * DY;
      const float* wrow = &wt[row * 128];
      const float* yp   = y2 + (size_t)mt * DY + dy;
      float s = 0.0f;
      for (int c = 0; c < 128; ++c) s += wrow[c] * yp[(size_t)c * DY];
      pacc[idx] += s;
    }
    if (tid < 64) {
      const float* wrow = &wt[tid * 128];
      float s = 0.0f;
      for (int c = 0; c < 128; ++c) s += wrow[c];
      rsum[tid] += s;
    }
    __syncthreads();
  }

  for (int idx = tid; idx < 64 * DY; idx += 128) {
    const int row = idx / DY, dy = idx - row * DY;
    out[(size_t)(row_blk + row) * DY + dy] = pacc[idx] / rsum[row];
  }
}

// ------------------------------ launcher -----------------------------------
extern "C" void kernel_launch(void* const* d_in, const int* in_sizes, int n_in,
                              void* d_out, int out_size, void* d_ws, size_t ws_size,
                              hipStream_t stream) {
  const float* x   = (const float*)d_in[0];  // Bn x D_IN
  const float* W   = (const float*)d_in[1];  // D_IN x D_ENC
  const float* b   = (const float*)d_in[2];  // D_ENC
  const float* cx  = (const float*)d_in[3];  // Mn x D_ENC
  const float* cy  = (const float*)d_in[4];  // Mn x DY
  const float* cw  = (const float*)d_in[5];  // Mn
  const float* sig = (const float*)d_in[6];  // 1
  float* out = (float*)d_out;

  char* p = (char*)d_ws;
  half_t* x_h   = (half_t*)p; p += (size_t)Bn * D_IN * sizeof(half_t);      // 8 MB
  half_t* wt_h  = (half_t*)p; p += (size_t)D_ENC * D_IN * sizeof(half_t);   // 1 MB
  half_t* cx_h  = (half_t*)p; p += (size_t)Mn * D_ENC * sizeof(half_t);     // 4 MB
  half_t* enc_h = (half_t*)p; p += (size_t)Bn * D_ENC * sizeof(half_t);     // 4 MB
  float* enc_n2 = (float*)p;  p += (size_t)Bn * sizeof(float);
  float* c_n2   = (float*)p;  p += (size_t)Mn * sizeof(float);
  float* y2     = (float*)p;  p += (size_t)Mn * DY * sizeof(float);

  // -- prep: f16 conversions + norms + normalized y^2
  cvt_f16_kernel<<<(Bn * D_IN + 255) / 256, 256, 0, stream>>>(x, x_h, Bn * D_IN);
  cvt_f16_T_kernel<<<(D_IN * D_ENC + 255) / 256, 256, 0, stream>>>(W, wt_h, D_IN, D_ENC);
  cvt_f16_kernel<<<(Mn * D_ENC + 255) / 256, 256, 0, stream>>>(cx, cx_h, Mn * D_ENC);
  row_norm2_f32_kernel<<<Mn, 32, 0, stream>>>(cx, c_n2, D_ENC);
  prep_y2_kernel<<<(Mn + 255) / 256, 256, 0, stream>>>(cy, y2, Mn);

  // -- GEMM1: enc = relu(x @ W + b)
  gemm_enc_kernel<<<dim3(Bn / 128, D_ENC / 128), 256, 0, stream>>>(x_h, wt_h, b, enc_h);
  row_norm2_f16_kernel<<<Bn, 32, 0, stream>>>(enc_h, enc_n2, D_ENC);

  // -- GEMM2 fused RBF + QKM mixture + dm2discrete
  gemm_qkm_kernel<<<Bn / 64, 128, 0, stream>>>(enc_h, cx_h, enc_n2, c_n2, cw,
                                               y2, sig, out);
}
